// AttentionHead_2439541424557
// MI455X (gfx1250) — matrix operand order repaired and checked by reading, not compiled, not run on hardware
//
#include <hip/hip_runtime.h>
#include <stdint.h>

// ---- CDNA5 (gfx1250) wave32 WMMA types ----
typedef __attribute__((ext_vector_type(16))) __bf16 v16bf;
typedef __attribute__((ext_vector_type(8)))  float  v8f;

#define NB 4
#define NS 4096
#define ND 256

__device__ __forceinline__ v8f wmma_bf16(v16bf a, v16bf b, v8f c) {
  // D = A(16x32 bf16) * B(32x16 bf16) + C(16x16 f32)
  return __builtin_amdgcn_wmma_f32_16x16x32_bf16(false, a, false, b, (short)0, c,
                                                 false, false);
}

#define PACK4(v, base, f)                                                      \
  do {                                                                         \
    (v)[(base) + 0] = (__bf16)(f).x;                                           \
    (v)[(base) + 1] = (__bf16)(f).y;                                           \
    (v)[(base) + 2] = (__bf16)(f).z;                                           \
    (v)[(base) + 3] = (__bf16)(f).w;                                           \
  } while (0)

// B-fragment (32x16, K=d-dim) of a weight matrix W[out=256][in=256] for
// y = x @ W^T : B[k][n] = W[n0+n][kc*32 + k].  lane: n = lane&15,
// elements e -> k = kbase + e (contiguous in memory -> 4x float4 loads).
__device__ __forceinline__ v16bf load_w_frag(const float* __restrict__ W,
                                             int n0, int nl, int kc, int kbase) {
  const float* wp = W + (size_t)(n0 + nl) * ND + kc * 32 + kbase;
  float4 g0 = *(const float4*)(wp + 0);
  float4 g1 = *(const float4*)(wp + 4);
  float4 g2 = *(const float4*)(wp + 8);
  float4 g3 = *(const float4*)(wp + 12);
  v16bf r;
  PACK4(r, 0, g0); PACK4(r, 4, g1); PACK4(r, 8, g2); PACK4(r, 12, g3);
  return r;
}

// ---------------------------------------------------------------------------
// Phase A: QKV projection.  One wave handles (b, s-tile of 16, e-chunk of 32).
// Emits Q as A-fragments, K as B-fragments, V transposed (Vt[d][s]) in bf16.
// ---------------------------------------------------------------------------
__global__ __launch_bounds__(128) void qkv_proj_kernel(
    const float* __restrict__ X,
    const float* __restrict__ Wq, const float* __restrict__ bq,
    const float* __restrict__ Wk, const float* __restrict__ bk,
    const float* __restrict__ Wv, const float* __restrict__ bv,
    __bf16* __restrict__ qfrag, __bf16* __restrict__ kfrag,
    __bf16* __restrict__ vt) {
  __shared__ __bf16 Qs[4][16][32];
  __shared__ __bf16 Ks[4][16][32];
  __shared__ __bf16 Vs[4][32][16];

  const int lane = threadIdx.x & 31;
  const int wid  = threadIdx.x >> 5;
  const int task = blockIdx.x * 4 + wid;   // NB * (NS/16) * (ND/32) = 8192
  const int ec = task & 7;                 // e-chunk (32 wide)
  const int st = (task >> 3) & 255;        // s-tile
  const int b  = task >> 11;

  const int nl    = lane & 15;
  const int hi    = lane >> 4;
  const int hi8   = hi * 8;
  const int kbase = hi * 16;

  // A-matrix row for this lane (both halves read the same row m = nl).
  const float* Xrow = X + ((size_t)b * NS + st * 16 + nl) * ND;

  v8f cq0 = {}, cq1 = {}, ck0 = {}, ck1 = {}, cv0 = {}, cv1 = {};

#pragma unroll
  for (int kc = 0; kc < 8; ++kc) {
    // A fragment of X (16x32 bf16): e -> k = kbase+e (contiguous)
    const float* xp = Xrow + kc * 32;
    float4 f0 = *(const float4*)(xp + kbase);
    float4 f1 = *(const float4*)(xp + kbase + 4);
    float4 f2 = *(const float4*)(xp + kbase + 8);
    float4 f3 = *(const float4*)(xp + kbase + 12);
    v16bf a;
    PACK4(a, 0, f0); PACK4(a, 4, f1); PACK4(a, 8, f2); PACK4(a, 12, f3);

    const int e0 = ec * 32;
    cq0 = wmma_bf16(a, load_w_frag(Wq, e0,      nl, kc, kbase), cq0);
    cq1 = wmma_bf16(a, load_w_frag(Wq, e0 + 16, nl, kc, kbase), cq1);
    ck0 = wmma_bf16(a, load_w_frag(Wk, e0,      nl, kc, kbase), ck0);
    ck1 = wmma_bf16(a, load_w_frag(Wk, e0 + 16, nl, kc, kbase), ck1);
    cv0 = wmma_bf16(a, load_w_frag(Wv, e0,      nl, kc, kbase), cv0);
    cv1 = wmma_bf16(a, load_w_frag(Wv, e0 + 16, nl, kc, kbase), cv1);
  }

  const int e0 = ec * 32;
  const float bq0 = bq[e0 + nl], bq1 = bq[e0 + 16 + nl];
  const float bk0 = bk[e0 + nl], bk1 = bk[e0 + 16 + nl];
  const float bv0 = bv[e0 + nl], bv1 = bv[e0 + 16 + nl];

  // D-layout: VGPR r holds (row = r + 8*hi, col = nl).
#pragma unroll
  for (int r = 0; r < 8; ++r) {
    const int row = r + hi8;
    Qs[wid][row][nl]      = (__bf16)(cq0[r] + bq0);
    Qs[wid][row][16 + nl] = (__bf16)(cq1[r] + bq1);
    Ks[wid][row][nl]      = (__bf16)(ck0[r] + bk0);
    Ks[wid][row][16 + nl] = (__bf16)(ck1[r] + bk1);
    Vs[wid][nl][row]      = (__bf16)(cv0[r] + bv0);   // transposed store
    Vs[wid][16 + nl][row] = (__bf16)(cv1[r] + bv1);
  }
  asm volatile("s_wait_dscnt 0" ::: "memory");  // in-wave cross-lane LDS sync

  // Q as A-fragment (k-dim = e): lane row m = nl.
  v16bf qa;
  *((uint4*)&qa)       = *(const uint4*)&Qs[wid][nl][kbase];
  *(((uint4*)&qa) + 1) = *(const uint4*)&Qs[wid][nl][kbase + 8];
  *(v16bf*)(qfrag + ((((size_t)b * 256 + st) * 8 + ec) * 32 + lane) * 16) = qa;

  // K as B-fragment (k-dim = e, n = s): lane col n = nl, elems k = kbase+e.
  v16bf kf = *(const v16bf*)&Ks[wid][nl][kbase];
  *(v16bf*)(kfrag + ((((size_t)b * 8 + ec) * 256 + st) * 32 + lane) * 16) = kf;

  // Vt[b][e][s]: lane writes e-row (e0 + lane), 16 contiguous s values.
  v16bf vf = *(const v16bf*)&Vs[wid][lane][0];
  *(v16bf*)(vt + ((size_t)(b * ND + e0 + lane)) * NS + st * 16) = vf;
}

// ---------------------------------------------------------------------------
// Phase B: flash attention, split-KV.  One wave per (b, q-tile, kv-split).
// NSPLIT=2 doubles wave count (latency hiding: softmax VALU of one wave
// overlaps the WMMA bursts of the other on the same SIMD).  NSPLIT=1 is the
// ws-constrained fallback that normalizes and writes the output directly.
// ---------------------------------------------------------------------------
template <int NSPLIT>
__global__ __launch_bounds__(128) void attn_kernel(
    const uint8_t* __restrict__ mask, const __bf16* __restrict__ qfrag,
    const __bf16* __restrict__ kfrag, const __bf16* __restrict__ vt,
    float* __restrict__ out, float* __restrict__ opart,
    float* __restrict__ ml) {
  __shared__ __bf16 Ps[4][16][32];

  const int lane  = threadIdx.x & 31;
  const int wid   = threadIdx.x >> 5;
  const int t     = blockIdx.x * 4 + wid;   // NB*(NS/16)*NSPLIT waves
  const int split = (NSPLIT > 1) ? (t & (NSPLIT - 1)) : 0;
  const int tt    = t / NSPLIT;
  const int b     = tt >> 8;
  const int qt    = tt & 255;
  const int q0    = qt * 16;

  const int nl    = lane & 15;
  const int hi    = lane >> 4;
  const int hi8   = hi * 8;
  const int kbase = hi * 16;

  v16bf qa[8];
#pragma unroll
  for (int d = 0; d < 8; ++d)
    qa[d] = *(const v16bf*)(qfrag +
                            ((((size_t)b * 256 + qt) * 8 + d) * 32 + lane) * 16);

  float mrow[8], lrow[8];
  v8f acc[16];
#pragma unroll
  for (int r = 0; r < 8; ++r) { mrow[r] = -3.0e38f; lrow[r] = 0.0f; }
#pragma unroll
  for (int n = 0; n < 16; ++n) { v8f z = {}; acc[n] = z; }

  const float scl = 0.015625f;  // 1/sqrt(S) = 1/64 (reference quirk: sqrt(seq))

  const int j0 = split * (128 / NSPLIT);
  const int j1 = j0 + 128 / NSPLIT;
  for (int j = j0; j < j1; ++j) {
    v8f c0 = {}, c1 = {};
#pragma unroll
    for (int d = 0; d < 8; ++d) {
      v16bf k0 = *(const v16bf*)(kfrag +
          ((((size_t)b * 8 + d) * 256 + 2 * j) * 32 + lane) * 16);
      v16bf k1 = *(const v16bf*)(kfrag +
          ((((size_t)b * 8 + d) * 256 + 2 * j + 1) * 32 + lane) * 16);
      c0 = wmma_bf16(qa[d], k0, c0);
      c1 = wmma_bf16(qa[d], k1, c1);
    }
    if (j + 1 < j1) {  // prefetch next step's first K fragments (L2-resident)
      __builtin_prefetch(kfrag + (((size_t)b * 8 * 256 + 2 * j + 2) * 32 + lane) * 16, 0, 1);
      __builtin_prefetch(kfrag + (((size_t)b * 8 * 256 + 2 * j + 3) * 32 + lane) * 16, 0, 1);
    }

    // Scale + mask (mask==true -> -1e9, matching jnp.where(mask, -1e9, s)).
    float s0r[8], s1r[8];
#pragma unroll
    for (int r = 0; r < 8; ++r) {
      const uint8_t* mp =
          mask + ((size_t)b * NS + q0 + r + hi8) * NS + 32 * j;
      s0r[r] = mp[nl]      ? -1.0e9f : c0[r] * scl;
      s1r[r] = mp[16 + nl] ? -1.0e9f : c1[r] * scl;
    }

    // Online softmax: row stats live replicated across the 16-lane half that
    // owns rows (hi*8 .. hi*8+7); xor-shuffles <= 8 stay inside the half.
    float p0[8], p1[8], alphas[8];
#pragma unroll
    for (int r = 0; r < 8; ++r) {
      float mx = fmaxf(s0r[r], s1r[r]);
#pragma unroll
      for (int off = 8; off >= 1; off >>= 1)
        mx = fmaxf(mx, __shfl_xor(mx, off, 32));
      const float mnew = fmaxf(mrow[r], mx);
      const float al   = __expf(mrow[r] - mnew);
      const float e0   = __expf(s0r[r] - mnew);
      const float e1   = __expf(s1r[r] - mnew);
      float sum = e0 + e1;
#pragma unroll
      for (int off = 8; off >= 1; off >>= 1)
        sum += __shfl_xor(sum, off, 32);
      lrow[r] = lrow[r] * al + sum;
      mrow[r] = mnew;
      alphas[r] = al;
      p0[r] = e0;
      p1[r] = e1;
    }

#pragma unroll
    for (int n = 0; n < 16; ++n)
#pragma unroll
      for (int r = 0; r < 8; ++r) acc[n][r] *= alphas[r];

    // Repack P (D-layout) into an A-fragment (16x32 bf16) through LDS.
#pragma unroll
    for (int r = 0; r < 8; ++r) {
      Ps[wid][r + hi8][nl]      = (__bf16)p0[r];
      Ps[wid][r + hi8][16 + nl] = (__bf16)p1[r];
    }
    asm volatile("s_wait_dscnt 0" ::: "memory");
    v16bf pa;
    *((uint4*)&pa)       = *(const uint4*)&Ps[wid][nl][kbase];
    *(((uint4*)&pa) + 1) = *(const uint4*)&Ps[wid][nl][kbase + 8];

    // context += P (16x32) @ V (32x16 per d-chunk); Vt gives contiguous loads.
#pragma unroll
    for (int n = 0; n < 16; ++n) {
      const __bf16* vp =
          vt + ((size_t)(b * ND + n * 16 + nl)) * NS + 32 * j + kbase;
      v16bf vb = *(const v16bf*)vp;
      acc[n] = wmma_bf16(pa, vb, acc[n]);
    }
  }

  if (NSPLIT == 1) {
    // Single pass: normalize and write the final output.
    float rinv[8];
#pragma unroll
    for (int r = 0; r < 8; ++r) rinv[r] = 1.0f / lrow[r];
#pragma unroll
    for (int n = 0; n < 16; ++n)
#pragma unroll
      for (int r = 0; r < 8; ++r)
        out[((size_t)b * NS + q0 + r + hi8) * ND + n * 16 + nl] =
            acc[n][r] * rinv[r];
  } else {
    // Split pass: store unnormalized accumulator + per-row (m, l).
    const size_t rowbase = (size_t)(split * NB + b) * NS + q0;
#pragma unroll
    for (int n = 0; n < 16; ++n)
#pragma unroll
      for (int r = 0; r < 8; ++r)
        opart[(rowbase + r + hi8) * ND + n * 16 + nl] = acc[n][r];
    if (nl == 0) {  // stats replicated across the half; one writer per half
#pragma unroll
      for (int r = 0; r < 8; ++r) {
        const size_t ridx = (rowbase + r + hi8) * 2;
        ml[ridx]     = mrow[r];
        ml[ridx + 1] = lrow[r];
      }
    }
  }
}

// ---------------------------------------------------------------------------
// Merge the two kv-splits: log-sum-exp combine, then normalize.
// ---------------------------------------------------------------------------
__global__ __launch_bounds__(256) void merge_kernel(
    const float* __restrict__ opart, const float* __restrict__ ml,
    float* __restrict__ out) {
  const size_t gid = (size_t)blockIdx.x * 256 + threadIdx.x;  // NB*NS*ND
  const int    d   = (int)(gid & (ND - 1));
  const size_t row = gid >> 8;                 // b*NS + s
  const size_t rows = (size_t)NB * NS;

  const float m1 = ml[row * 2],          l1 = ml[row * 2 + 1];
  const float m2 = ml[(rows + row) * 2], l2 = ml[(rows + row) * 2 + 1];
  const float mm = fmaxf(m1, m2);
  const float e1 = __expf(m1 - mm);
  const float e2 = __expf(m2 - mm);
  const float rinv = 1.0f / (e1 * l1 + e2 * l2);

  const float o1 = opart[row * ND + d];
  const float o2 = opart[rows * ND + row * ND + d];
  out[row * ND + d] = (e1 * o1 + e2 * o2) * rinv;
}

// ---------------------------------------------------------------------------
extern "C" void kernel_launch(void* const* d_in, const int* in_sizes, int n_in,
                              void* d_out, int out_size, void* d_ws,
                              size_t ws_size, hipStream_t stream) {
  const float*   X    = (const float*)d_in[0];
  const uint8_t* mask = (const uint8_t*)d_in[1];  // jnp bool -> 1 byte
  const float*   Wq   = (const float*)d_in[2];
  const float*   bq   = (const float*)d_in[3];
  const float*   Wk   = (const float*)d_in[4];
  const float*   bk   = (const float*)d_in[5];
  const float*   Wv   = (const float*)d_in[6];
  const float*   bv   = (const float*)d_in[7];
  float*         out  = (float*)d_out;

  // Workspace partition: Q frags 8MB | K frags 8MB | Vt 8MB | [O parts 32MB | ml 256KB]
  const size_t fragElems = (size_t)NB * (NS / 16) * (ND / 32) * 32 * 16;
  __bf16* qfrag = (__bf16*)d_ws;
  __bf16* kfrag = qfrag + fragElems;
  __bf16* vt    = kfrag + fragElems;
  float*  opart = (float*)(vt + fragElems);                // 2*NB*NS*ND f32
  float*  ml    = opart + (size_t)2 * NB * NS * ND;        // 2*NB*NS*2  f32

  const size_t needSplit = fragElems * 3 * sizeof(__bf16) +
                           (size_t)2 * NB * NS * ND * sizeof(float) +
                           (size_t)2 * NB * NS * 2 * sizeof(float);

  const int tasksA = NB * (NS / 16) * (ND / 32);  // 8192 waves
  qkv_proj_kernel<<<tasksA / 4, 128, 0, stream>>>(X, Wq, bq, Wk, bk, Wv, bv,
                                                  qfrag, kfrag, vt);

  if (ws_size >= needSplit) {
    const int tasksB = NB * (NS / 16) * 2;        // 2048 waves (split-KV)
    attn_kernel<2><<<tasksB / 4, 128, 0, stream>>>(mask, qfrag, kfrag, vt, out,
                                                   opart, ml);
    merge_kernel<<<(NB * NS * ND) / 256, 256, 0, stream>>>(opart, ml, out);
  } else {
    const int tasksB = NB * (NS / 16);            // 1024 waves (single pass)
    attn_kernel<1><<<tasksB / 4, 128, 0, stream>>>(mask, qfrag, kfrag, vt, out,
                                                   opart, ml);
  }
}
